// InteractionDecoder_68504728371708
// MI455X (gfx1250) — matrix operand order repaired and checked by reading, not compiled, not run on hardware
//
#include <hip/hip_runtime.h>

typedef _Float16 f16;
typedef _Float16 v16h __attribute__((ext_vector_type(16)));
typedef _Float16 v8h  __attribute__((ext_vector_type(8)));
typedef float    v8f  __attribute__((ext_vector_type(8)));

// ---------------- K0: zero segment-sum buffers ----------------
__global__ void k_zero(float* p, int n) {
    int i = blockIdx.x * blockDim.x + threadIdx.x;
    if (i < n) p[i] = 0.f;
}

// ---------------- K1a: swizzle W1 -> WMMA f16 B-fragment layout ----------------
// Fragment f = (mlp*8+nt)*4+kb, 32 lanes x 16 halves each (1 KB per fragment).
// B (32x16 KxN, f16): lanes 0-15 hold K=0..7,16..23 (halves 0..15), lanes 16-31 hold
// K=8..15,24..31; N = lane%16.  K_global = kb*32 + chunk, N_global = nt*16 + lane%16.
__global__ void k_prep_b(f16* __restrict__ bswz,
                         const float* __restrict__ w10, const float* __restrict__ w11,
                         const float* __restrict__ w12, const float* __restrict__ w13) {
    int t = blockIdx.x * blockDim.x + threadIdx.x;          // 0..65535
    int h   = t & 15;
    int L   = (t >> 4) & 31;
    int kb  = (t >> 9) & 3;
    int nt  = (t >> 11) & 7;
    int mlp = t >> 14;
    const float* w1 = (mlp == 0) ? w10 : (mlp == 1) ? w11 : (mlp == 2) ? w12 : w13;
    int K  = kb * 32 + ((h < 8) ? h : 8 + h) + ((L < 16) ? 0 : 8);
    int Nc = nt * 16 + (L & 15);
    bswz[t] = (f16)w1[K * 128 + Nc];
}

// ---------------- K1b: swizzle W2 (padded to 16 cols) -> B-fragment layout ----
// Fragment f = mlp*4 + kb, 512 halves each. Columns >= dout are zero.
__global__ void k_prep_b2(f16* __restrict__ bswz2,
                          const float* __restrict__ w20, const float* __restrict__ w21,
                          const float* __restrict__ w22, const float* __restrict__ w23) {
    int t = blockIdx.x * blockDim.x + threadIdx.x;          // 0..8191
    int h   = t & 15;
    int L   = (t >> 4) & 31;
    int kb  = (t >> 9) & 3;
    int mlp = t >> 11;
    const float* w2 = (mlp == 0) ? w20 : (mlp == 1) ? w21 : (mlp == 2) ? w22 : w23;
    int dout = (mlp == 2) ? 1 : 3;
    int K = kb * 32 + ((h < 8) ? h : 8 + h) + ((L < 16) ? 0 : 8);
    int n = L & 15;
    f16 v = (f16)0.f;
    if (n < dout) v = (f16)w2[K * dout + n];
    bswz2[t] = v;
}

// ---------------- K2: fused WMMA MLPs + edge vector assembly ----------------
struct GArgs {
    const float* x;          // [E,128]
    const f16*   bswz;       // swizzled W1 fragments (layer 1)
    const f16*   bswz2;      // swizzled padded W2 fragments (layer 2)
    const float *b1_0, *b1_1, *b1_2, *b1_3;
    const float *b2_0, *b2_1, *b2_2, *b2_3;
    const float *va, *vb, *vc;                // [E,3]
    const int   *eidx, *eattr, *ntype;
    float *fw, *aw, *dw, *lw, *sum_in;
    int E;
};

__global__ __launch_bounds__(256)
void k_edge_gemm(GArgs A) {
    __shared__ __align__(16) f16   xs[64 * 128];     // x tile in f16     (16 KB)
    __shared__ __align__(16) f16   hbuf[64 * 512];   // post-relu hidden  (64 KB)
    __shared__ float coeffs[64 * 10];                // layer-2 outputs   (2.5 KB)

    const int t  = threadIdx.x;
    const int e0 = blockIdx.x * 64;

    // prefetch next tile's activations (global_prefetch_b8)
    if ((e0 + 64) < A.E && t == 0)
        __builtin_prefetch(A.x + (size_t)(e0 + 64) * 128, 0, 1);

    // ---- Phase A: cooperative x tile load -> f16 LDS (row-major [64][128]) ----
    {
        int lr = t >> 2;          // 0..63 (edge row)
        int c0 = (t & 3) * 32;    // 0,32,64,96
        int e  = e0 + lr;
        float4 u[8];
#pragma unroll
        for (int i = 0; i < 8; ++i) u[i] = make_float4(0.f, 0.f, 0.f, 0.f);
        if (e < A.E) {
            const float4* px = (const float4*)(A.x + (size_t)e * 128 + c0);
#pragma unroll
            for (int i = 0; i < 8; ++i) u[i] = px[i];
        }
        f16* dst = &xs[lr * 128 + c0];
#pragma unroll
        for (int i = 0; i < 8; ++i) {
            dst[i*4+0] = (f16)u[i].x; dst[i*4+1] = (f16)u[i].y;
            dst[i*4+2] = (f16)u[i].z; dst[i*4+3] = (f16)u[i].w;
        }
    }
    __syncthreads();

    const int lane = t & 31, wave = t >> 5;
    const int row16 = lane & 15;
    const int kOff  = (lane < 16) ? 0 : 8;   // A 16x32 f16 layout
    const int mHi   = (lane >> 4) * 8;       // D layout: M = r + 8*(lane/16)

    // ---- Phase B: layer-1 GEMM via WMMA, dual-accumulator M-tile pairs ----
    {
        const int mlp = wave >> 1, ntBase = (wave & 1) * 4;
        const float* b1p = (mlp == 0) ? A.b1_0 : (mlp == 1) ? A.b1_1 : (mlp == 2) ? A.b1_2 : A.b1_3;

        for (int nti = 0; nti < 4; ++nti) {
            const int nt = ntBase + nti;
            // load B fragments once, reuse across the 4 M-tiles
            v16h b[4];
            const f16* fb = A.bswz + (size_t)((mlp * 8 + nt) * 4) * 512 + lane * 16;
#pragma unroll
            for (int kb = 0; kb < 4; ++kb) {
                const v8h* pb = (const v8h*)(fb + kb * 512);
                v8h blo = pb[0], bhi = pb[1];
                b[kb] = __builtin_shufflevector(blo, bhi, 0,1,2,3,4,5,6,7,8,9,10,11,12,13,14,15);
            }
            const float bias = b1p[nt * 16 + row16];
            const int   col  = mlp * 128 + nt * 16 + row16;

            for (int mp = 0; mp < 2; ++mp) {
                const int mtA = mp * 2, mtB = mtA + 1;
                v16h a0[4], a1[4];
#pragma unroll
                for (int kb = 0; kb < 4; ++kb) {
                    const int r0 = (mtA * 16 + row16) * 128 + kb * 32 + kOff;
                    const int r1 = (mtB * 16 + row16) * 128 + kb * 32 + kOff;
                    v8h l0 = *(const v8h*)&xs[r0];
                    v8h h0 = *(const v8h*)&xs[r0 + 16];
                    v8h l1 = *(const v8h*)&xs[r1];
                    v8h h1 = *(const v8h*)&xs[r1 + 16];
                    a0[kb] = __builtin_shufflevector(l0, h0, 0,1,2,3,4,5,6,7,8,9,10,11,12,13,14,15);
                    a1[kb] = __builtin_shufflevector(l1, h1, 0,1,2,3,4,5,6,7,8,9,10,11,12,13,14,15);
                }
                v8f acc0 = {0.f, 0.f, 0.f, 0.f, 0.f, 0.f, 0.f, 0.f};
                v8f acc1 = {0.f, 0.f, 0.f, 0.f, 0.f, 0.f, 0.f, 0.f};
#pragma unroll
                for (int kb = 0; kb < 4; ++kb) {
                    acc0 = __builtin_amdgcn_wmma_f32_16x16x32_f16(
                            false, a0[kb], false, b[kb], (short)0, acc0, false, false);
                    acc1 = __builtin_amdgcn_wmma_f32_16x16x32_f16(
                            false, a1[kb], false, b[kb], (short)0, acc1, false, false);
                }
#pragma unroll
                for (int r = 0; r < 8; ++r) {
                    float h0 = acc0[r] + bias; h0 = h0 > 0.f ? h0 : 0.f;
                    float h1 = acc1[r] + bias; h1 = h1 > 0.f ? h1 : 0.f;
                    hbuf[(mtA * 16 + r + mHi) * 512 + col] = (f16)h0;
                    hbuf[(mtB * 16 + r + mHi) * 512 + col] = (f16)h1;
                }
            }
        }
    }
    __syncthreads();

    // ---- Phase C: layer 2 via WMMA (hidden[16x128] @ W2pad[128x16]) ----
    // wave -> mlp2 = wave>>1, M-tile pair = (wave&1)*2 .. +1 ; shares B frags.
    {
        const int mlp2 = wave >> 1;
        const int mtA  = (wave & 1) * 2, mtB = mtA + 1;
        const int jbase = (mlp2 == 0) ? 0 : (mlp2 == 1) ? 3 : (mlp2 == 2) ? 6 : 7;
        const int dout  = (mlp2 == 2) ? 1 : 3;

        v16h b[4];
#pragma unroll
        for (int kb = 0; kb < 4; ++kb) {
            const v8h* pb = (const v8h*)(A.bswz2 + (size_t)(mlp2 * 4 + kb) * 512 + lane * 16);
            v8h blo = pb[0], bhi = pb[1];
            b[kb] = __builtin_shufflevector(blo, bhi, 0,1,2,3,4,5,6,7,8,9,10,11,12,13,14,15);
        }

        v8f acc0 = {0.f, 0.f, 0.f, 0.f, 0.f, 0.f, 0.f, 0.f};
        v8f acc1 = {0.f, 0.f, 0.f, 0.f, 0.f, 0.f, 0.f, 0.f};
#pragma unroll
        for (int kb = 0; kb < 4; ++kb) {
            const int r0 = (mtA * 16 + row16) * 512 + mlp2 * 128 + kb * 32 + kOff;
            const int r1 = (mtB * 16 + row16) * 512 + mlp2 * 128 + kb * 32 + kOff;
            v8h l0 = *(const v8h*)&hbuf[r0];
            v8h h0 = *(const v8h*)&hbuf[r0 + 16];
            v8h l1 = *(const v8h*)&hbuf[r1];
            v8h h1 = *(const v8h*)&hbuf[r1 + 16];
            v16h a0 = __builtin_shufflevector(l0, h0, 0,1,2,3,4,5,6,7,8,9,10,11,12,13,14,15);
            v16h a1 = __builtin_shufflevector(l1, h1, 0,1,2,3,4,5,6,7,8,9,10,11,12,13,14,15);
            acc0 = __builtin_amdgcn_wmma_f32_16x16x32_f16(
                    false, a0, false, b[kb], (short)0, acc0, false, false);
            acc1 = __builtin_amdgcn_wmma_f32_16x16x32_f16(
                    false, a1, false, b[kb], (short)0, acc1, false, false);
        }
        if (row16 < dout) {
#pragma unroll
            for (int r = 0; r < 8; ++r) {
                coeffs[(mtA * 16 + r + mHi) * 10 + jbase + row16] = acc0[r];
                coeffs[(mtB * 16 + r + mHi) * 10 + jbase + row16] = acc1[r];
            }
        }
    }
    __syncthreads();

    // ---- Phase D: raw fij/aij/dxij/lambda + receiver segment sums ----
    if (t < 64 && (e0 + t) < A.E) {
        const int e = e0 + t;
        const float* c = &coeffs[t * 10];
        float c0 = c[0] + A.b2_0[0], c1 = c[1] + A.b2_0[1], c2 = c[2] + A.b2_0[2];
        float c3 = c[3] + A.b2_1[0], c4 = c[4] + A.b2_1[1], c5 = c[5] + A.b2_1[2];
        float lam = c[6] + A.b2_2[0];
        float c7 = c[7] + A.b2_3[0], c8 = c[8] + A.b2_3[1], c9 = c[9] + A.b2_3[2];

        float vax = A.va[e*3], vay = A.va[e*3+1], vaz = A.va[e*3+2];
        float vbx = A.vb[e*3], vby = A.vb[e*3+1], vbz = A.vb[e*3+2];
        float vcx = A.vc[e*3], vcy = A.vc[e*3+1], vcz = A.vc[e*3+2];

        float fx = c0*vax + c1*vbx + c2*vcx;
        float fy = c0*vay + c1*vby + c2*vcy;
        float fz = c0*vaz + c1*vbz + c2*vcz;
        float ax = c3*vax + c4*vbx + c5*vcx;
        float ay = c3*vay + c4*vby + c5*vcy;
        float az = c3*vaz + c4*vbz + c5*vcz;
        float dx = c7*vax + c8*vbx + c9*vcx;
        float dy = c7*vay + c8*vby + c9*vcy;
        float dz = c7*vaz + c8*vbz + c9*vcz;

        A.fw[e*3] = fx; A.fw[e*3+1] = fy; A.fw[e*3+2] = fz;
        A.aw[e*3] = ax; A.aw[e*3+1] = ay; A.aw[e*3+2] = az;
        A.dw[e*3] = dx; A.dw[e*3+1] = dy; A.dw[e*3+2] = dz;
        A.lw[e]   = lam;

        const int virt = (A.eattr[e] == -1);
        const int rcv  = A.eidx[A.E + e];
        if (virt && (A.ntype[rcv * 2 + 1] == -1)) {
            float* s_ = A.sum_in + (size_t)rcv * 10;
            atomicAdd(s_+0, fx); atomicAdd(s_+1, fy); atomicAdd(s_+2, fz);
            atomicAdd(s_+3, ax); atomicAdd(s_+4, ay); atomicAdd(s_+5, az);
            atomicAdd(s_+6, dx); atomicAdd(s_+7, dy); atomicAdd(s_+8, dz);
            atomicAdd(s_+9, 1.0f);
        }
    }
}

// ---------------- K3: apply receiver-mean removal, accumulate sender sums ----------------
__global__ void k_mean_in(const int* __restrict__ eidx, const int* __restrict__ eattr,
                          const int* __restrict__ ntype, const float* __restrict__ sum_in,
                          float* __restrict__ sum_out,
                          float* fw, float* aw, float* dw, int E) {
    int e = blockIdx.x * blockDim.x + threadIdx.x;
    if (e >= E) return;
    if (eattr[e] != -1) return;                       // not virtual -> both masks false
    int snd = eidx[e], rcv = eidx[E + e];
    int gr = (ntype[rcv * 2 + 1] == -1);
    int gs = (ntype[snd * 2 + 1] == -1);
    if (!gr && !gs) return;

    float fx = fw[e*3], fy = fw[e*3+1], fz = fw[e*3+2];
    float ax = aw[e*3], ay = aw[e*3+1], az = aw[e*3+2];
    float dx = dw[e*3], dy = dw[e*3+1], dz = dw[e*3+2];

    if (gr) {
        const float* s_ = sum_in + (size_t)rcv * 10;
        float inv = 1.f / fmaxf(s_[9], 1.f);
        fx -= s_[0]*inv; fy -= s_[1]*inv; fz -= s_[2]*inv;
        ax -= s_[3]*inv; ay -= s_[4]*inv; az -= s_[5]*inv;
        dx -= s_[6]*inv; dy -= s_[7]*inv; dz -= s_[8]*inv;
        fw[e*3] = fx; fw[e*3+1] = fy; fw[e*3+2] = fz;
        aw[e*3] = ax; aw[e*3+1] = ay; aw[e*3+2] = az;
        dw[e*3] = dx; dw[e*3+1] = dy; dw[e*3+2] = dz;
    }
    if (gs) {
        float* o = sum_out + (size_t)snd * 10;
        atomicAdd(o+0, fx); atomicAdd(o+1, fy); atomicAdd(o+2, fz);
        atomicAdd(o+3, ax); atomicAdd(o+4, ay); atomicAdd(o+5, az);
        atomicAdd(o+6, dx); atomicAdd(o+7, dy); atomicAdd(o+8, dz);
        atomicAdd(o+9, 1.0f);
    }
}

// ---------------- K4: sender-mean removal, torque, outputs ----------------
__global__ void k_final(const int* __restrict__ eidx, const int* __restrict__ eattr,
                        const int* __restrict__ ntype, const float* __restrict__ sum_out,
                        const float* __restrict__ fw, const float* __restrict__ aw,
                        const float* __restrict__ dw, const float* __restrict__ lw,
                        const float* __restrict__ spos, const float* __restrict__ rpos,
                        const float* __restrict__ wn, float* __restrict__ out, int E) {
    int e = blockIdx.x * blockDim.x + threadIdx.x;
    if (e >= E) return;
    int snd = eidx[e], rcv = eidx[E + e];

    float fx = fw[e*3], fy = fw[e*3+1], fz = fw[e*3+2];
    float ax = aw[e*3], ay = aw[e*3+1], az = aw[e*3+2];
    float dx = dw[e*3], dy = dw[e*3+1], dz = dw[e*3+2];

    if ((eattr[e] == -1) && (ntype[snd * 2 + 1] == -1)) {
        const float* s_ = sum_out + (size_t)snd * 10;
        float inv = 1.f / fmaxf(s_[9], 1.f);
        fx -= s_[0]*inv; fy -= s_[1]*inv; fz -= s_[2]*inv;
        ax -= s_[3]*inv; ay -= s_[4]*inv; az -= s_[5]*inv;
        dx -= s_[6]*inv; dy -= s_[7]*inv; dz -= s_[8]*inv;
    }

    float lam = lw[e];
    float ws_ = wn[snd], wr_ = wn[rcv];
    float sx = spos[e*3], sy = spos[e*3+1], sz = spos[e*3+2];
    float rx = rpos[e*3], ry = rpos[e*3+1], rz = rpos[e*3+2];
    float invw = 1.f / (ws_ + wr_);
    float lx = rx - (ws_*sx + wr_*rx) * invw;
    float ly = ry - (ws_*sy + wr_*ry) * invw;
    float lz = rz - (ws_*sz + wr_*rz) * invw;
    float gx = fx * lam, gy = fy * lam, gz = fz * lam;
    float tx = ax - (ly*gz - lz*gy);
    float ty = ay - (lz*gx - lx*gz);
    float tz = az - (lx*gy - ly*gx);

    out[e*3]   = fx; out[e*3+1]   = fy; out[e*3+2]   = fz;
    out[(size_t)E*3 + e*3]   = tx; out[(size_t)E*3 + e*3+1]   = ty; out[(size_t)E*3 + e*3+2]   = tz;
    out[(size_t)E*6 + e*3]   = dx; out[(size_t)E*6 + e*3+1]   = dy; out[(size_t)E*6 + e*3+2]   = dz;
}

extern "C" void kernel_launch(void* const* d_in, const int* in_sizes, int n_in,
                              void* d_out, int out_size, void* d_ws, size_t ws_size,
                              hipStream_t stream) {
    const int E = in_sizes[1];      // edge_attr: E*1
    const int N = in_sizes[9];      // w_nodes:   N*1

    const int*   eidx  = (const int*)d_in[0];
    const int*   eattr = (const int*)d_in[1];
    const int*   ntype = (const int*)d_in[2];
    const float* spos  = (const float*)d_in[3];
    const float* rpos  = (const float*)d_in[4];
    const float* va    = (const float*)d_in[5];
    const float* vb    = (const float*)d_in[6];
    const float* vc    = (const float*)d_in[7];
    const float* x     = (const float*)d_in[8];
    const float* wn    = (const float*)d_in[9];
    const float* W1[4] = {(const float*)d_in[10], (const float*)d_in[14],
                          (const float*)d_in[18], (const float*)d_in[22]};
    const float* B1[4] = {(const float*)d_in[11], (const float*)d_in[15],
                          (const float*)d_in[19], (const float*)d_in[23]};
    const float* W2[4] = {(const float*)d_in[12], (const float*)d_in[16],
                          (const float*)d_in[20], (const float*)d_in[24]};
    const float* B2[4] = {(const float*)d_in[13], (const float*)d_in[17],
                          (const float*)d_in[21], (const float*)d_in[25]};

    // ---- workspace layout ----
    char* ws = (char*)d_ws;
    f16*   bswz  = (f16*)ws;                   size_t off = (size_t)4 * 128 * 128 * sizeof(f16);
    f16*   bswz2 = (f16*)(ws + off);           off += (size_t)4 * 4 * 512 * sizeof(f16);
    float* fw    = (float*)(ws + off);         off += (size_t)E * 3 * sizeof(float);
    float* aw    = (float*)(ws + off);         off += (size_t)E * 3 * sizeof(float);
    float* dw    = (float*)(ws + off);         off += (size_t)E * 3 * sizeof(float);
    float* lw    = (float*)(ws + off);         off += (size_t)E * sizeof(float);
    float* sum_in  = (float*)(ws + off);       off += (size_t)N * 10 * sizeof(float);
    float* sum_out = (float*)(ws + off);       off += (size_t)N * 10 * sizeof(float);
    (void)ws_size; (void)n_in; (void)out_size;

    // K0: zero both segment-sum buffers (contiguous)
    {
        int n = N * 20;
        k_zero<<<(n + 255) / 256, 256, 0, stream>>>(sum_in, n);
    }
    // K1a/K1b: weight swizzles
    k_prep_b <<<256, 256, 0, stream>>>(bswz,  W1[0], W1[1], W1[2], W1[3]);
    k_prep_b2<<<32,  256, 0, stream>>>(bswz2, W2[0], W2[1], W2[2], W2[3]);

    // K2: fused WMMA MLPs + edge assembly + receiver sums (64 edges / block)
    GArgs A;
    A.x = x; A.bswz = bswz; A.bswz2 = bswz2;
    A.b1_0 = B1[0]; A.b1_1 = B1[1]; A.b1_2 = B1[2]; A.b1_3 = B1[3];
    A.b2_0 = B2[0]; A.b2_1 = B2[1]; A.b2_2 = B2[2]; A.b2_3 = B2[3];
    A.va = va; A.vb = vb; A.vc = vc;
    A.eidx = eidx; A.eattr = eattr; A.ntype = ntype;
    A.fw = fw; A.aw = aw; A.dw = dw; A.lw = lw; A.sum_in = sum_in;
    A.E = E;
    k_edge_gemm<<<(E + 63) / 64, 256, 0, stream>>>(A);

    // K3: receiver-mean removal + sender sums
    k_mean_in<<<(E + 255) / 256, 256, 0, stream>>>(eidx, eattr, ntype, sum_in, sum_out,
                                                   fw, aw, dw, E);
    // K4: sender-mean removal + torque + outputs
    k_final<<<(E + 255) / 256, 256, 0, stream>>>(eidx, eattr, ntype, sum_out,
                                                 fw, aw, dw, lw, spos, rpos, wn,
                                                 (float*)d_out, E);
}
